// GPT2_1984274891096
// MI455X (gfx1250) — matrix operand order repaired and checked by reading, not compiled, not run on hardware
//
#include <hip/hip_runtime.h>
#include <hip/hip_bf16.h>
#include <math.h>

// GPT-2 forward for MI455X (gfx1250, wave32, WMMA).
// Heavy math routed through v_wmma_f32_16x16x32_bf16 with vectorized
// (B128 global / b64-b128 LDS) staging on the tile-aligned fast path.

#define LAYERS  12
#define NHEADS  12
#define HID     768
#define VOCAB   50257
#define SEQ     1024
#define NBATCH  8
#define HD      64
#define MTOK    (NBATCH * SEQ)   // 8192 rows

typedef __attribute__((ext_vector_type(16))) __bf16 v16bf;
typedef __attribute__((ext_vector_type(8)))  __bf16 v8bf;
typedef __attribute__((ext_vector_type(4)))  __bf16 v4bf;
typedef __attribute__((ext_vector_type(8)))  float  v8f;

union FragBF { v16bf v; v8bf h[2]; };

// ---------------------------------------------------------------------------
// Embedding: x[b,s,:] = tok_emb[ids[b,s],:] + pos_emb[s,:]
// ---------------------------------------------------------------------------
__global__ __launch_bounds__(256)
void embed_kernel(const int* __restrict__ ids, const float* __restrict__ tok,
                  const float* __restrict__ pos, float* __restrict__ x)
{
    const long long bs = blockIdx.x;
    const int s = (int)(bs % SEQ);
    const long long id = ids[bs];
    for (int j = threadIdx.x; j < HID; j += 256)
        x[bs * HID + j] = tok[id * HID + j] + pos[(long long)s * HID + j];
}

// ---------------------------------------------------------------------------
// LayerNorm: one block per row (row length = HID)
// ---------------------------------------------------------------------------
__global__ __launch_bounds__(256)
void layernorm_kernel(const float* __restrict__ in, long long in_stride,
                      const float* __restrict__ g, const float* __restrict__ b,
                      float* __restrict__ out, long long out_stride, int n)
{
    const long long row = blockIdx.x;
    const float* x = in + row * in_stride;
    float* o = out + row * out_stride;
    __shared__ float s1[256];
    __shared__ float s2[256];
    float sum = 0.0f, sq = 0.0f;
    for (int j = threadIdx.x; j < n; j += 256) {
        float v = x[j];
        sum += v; sq += v * v;
    }
    s1[threadIdx.x] = sum; s2[threadIdx.x] = sq;
    __syncthreads();
    for (int st = 128; st > 0; st >>= 1) {
        if (threadIdx.x < st) {
            s1[threadIdx.x] += s1[threadIdx.x + st];
            s2[threadIdx.x] += s2[threadIdx.x + st];
        }
        __syncthreads();
    }
    const float mu   = s1[0] / (float)n;
    const float var  = s2[0] / (float)n - mu * mu;
    const float rstd = rsqrtf(var + 1e-5f);
    for (int j = threadIdx.x; j < n; j += 256)
        o[j] = (x[j] - mu) * rstd * g[j] + b[j];
}

// ---------------------------------------------------------------------------
// Causal softmax over scores rows. Row index = h*SEQ + t; masking & scaling
// fused (reference: where(mask,-1e4,scores)*scale, then softmax).
// ---------------------------------------------------------------------------
__global__ __launch_bounds__(256)
void softmax_kernel(float* __restrict__ scores, float scale)
{
    const long long row = blockIdx.x;
    const int t = (int)(row % SEQ);
    float* p = scores + row * (long long)SEQ;
    __shared__ float red[256];

    float mx = -3.0e38f;
    for (int j = threadIdx.x; j < SEQ; j += 256) {
        float v = (j > t) ? (-10000.0f * scale) : (p[j] * scale);
        mx = fmaxf(mx, v);
    }
    red[threadIdx.x] = mx; __syncthreads();
    for (int st = 128; st > 0; st >>= 1) {
        if (threadIdx.x < st) red[threadIdx.x] = fmaxf(red[threadIdx.x], red[threadIdx.x + st]);
        __syncthreads();
    }
    mx = red[0];
    __syncthreads();

    float sum = 0.0f;
    for (int j = threadIdx.x; j < SEQ; j += 256) {
        float v = (j > t) ? (-10000.0f * scale) : (p[j] * scale);
        float e = __expf(v - mx);
        p[j] = e;
        sum += e;
    }
    red[threadIdx.x] = sum; __syncthreads();
    for (int st = 128; st > 0; st >>= 1) {
        if (threadIdx.x < st) red[threadIdx.x] += red[threadIdx.x + st];
        __syncthreads();
    }
    const float inv = 1.0f / red[0];
    for (int j = threadIdx.x; j < SEQ; j += 256) p[j] *= inv;
}

// ---------------------------------------------------------------------------
// WMMA-tiled GEMM:  C = [res +] act(A @ B + bias)
//   A: M x K (f32, row-major, lda). B: TRANS_B ? (N x K, ldb) : (K x N, ldb).
//   Batched over blockIdx.z with element strides sA/sB/sR/sC.
// Block = 256 threads (8 waves), 128x64 output tile; wave grid 4(m) x 2(n),
// each wave owns a 32x32 sub-tile: 2 A-frags x 2 B-frags -> 4 wmma / k-step.
// Tile-aligned blocks take a fully vectorized, predication-free path
// (float4 global loads, b64 LDS stores, b128 LDS fragment reads).
// ---------------------------------------------------------------------------
#define PITCH 40   // bf16 elements per LDS row (8B aligned, odd bank-group)

template<bool TRANS_B, bool DO_GELU>
__global__ __launch_bounds__(256)
void gemm_wmma_kernel(const float* __restrict__ A,  long long sA, int lda,
                      const float* __restrict__ Bm, long long sB, int ldb,
                      const float* __restrict__ bias,
                      const float* __restrict__ res, long long sR, int ldr,
                      float* __restrict__ C,  long long sC, int ldc,
                      int M, int N, int K)
{
    A  += (long long)blockIdx.z * sA;
    Bm += (long long)blockIdx.z * sB;
    if (res) res += (long long)blockIdx.z * sR;
    C  += (long long)blockIdx.z * sC;

    const int bm  = blockIdx.y * 128;
    const int bn  = blockIdx.x * 64;
    const int tid = threadIdx.x;
    const int lane = tid & 31;
    const int w    = tid >> 5;
    const int wm   = (w >> 1) * 32;   // 0,32,64,96
    const int wn   = (w & 1) * 32;    // 0,32

    const bool interior = (bm + 128 <= M) && (bn + 64 <= N);

    __shared__ __bf16 As[128 * PITCH];  // [m][k]
    __shared__ __bf16 Bs[ 64 * PITCH];  // [n][k]

    v8f acc[2][2] = {};

    // ISA 7.12.2 fragment addressing (wave32):
    const int fr = lane & 15;                  // row (A) / col (B) in 16-tile
    const int kA = (lane < 16) ? 0 : 8;        // A: runs [kA..kA+7],[kA+16..kA+23]
    const int kB = (lane < 16) ? 0 : 8;        // B: runs [kB..kB+7],[kB+16..kB+23]

    for (int k0 = 0; k0 < K; k0 += 32) {
        __syncthreads();
        if (interior) {
            // ---- A tile 128x32: 4 x float4 per thread ----
            #pragma unroll
            for (int i = 0; i < 4; ++i) {
                int idx = i * 256 + tid;
                int r = idx >> 3, kq = (idx & 7) * 4;
                const float4 f = *(const float4*)(A + (size_t)(bm + r) * lda + (k0 + kq));
                v4bf p = { (__bf16)f.x, (__bf16)f.y, (__bf16)f.z, (__bf16)f.w };
                *(v4bf*)&As[r * PITCH + kq] = p;
            }
            // ---- B tile 64x32 (stored [n][k]) ----
            #pragma unroll
            for (int i = 0; i < 2; ++i) {
                int idx = i * 256 + tid;
                int n = idx & 63, kq = (idx >> 6) * 4;
                v4bf p;
                if (TRANS_B) {
                    const float4 f = *(const float4*)(Bm + (size_t)(bn + n) * ldb + (k0 + kq));
                    p = (v4bf){ (__bf16)f.x, (__bf16)f.y, (__bf16)f.z, (__bf16)f.w };
                } else {
                    const float f0 = Bm[(size_t)(k0 + kq + 0) * ldb + bn + n];
                    const float f1 = Bm[(size_t)(k0 + kq + 1) * ldb + bn + n];
                    const float f2 = Bm[(size_t)(k0 + kq + 2) * ldb + bn + n];
                    const float f3 = Bm[(size_t)(k0 + kq + 3) * ldb + bn + n];
                    p = (v4bf){ (__bf16)f0, (__bf16)f1, (__bf16)f2, (__bf16)f3 };
                }
                *(v4bf*)&Bs[n * PITCH + kq] = p;
            }
            // prefetch next k-tile (global_prefetch_b8)
            if (k0 + 32 < K) {
                __builtin_prefetch(A + (size_t)(bm + (tid >> 1)) * lda + (k0 + 32) + (tid & 1) * 16, 0, 1);
                if (TRANS_B)
                    __builtin_prefetch(Bm + (size_t)(bn + (tid & 63)) * ldb + (k0 + 32), 0, 1);
                else
                    __builtin_prefetch(Bm + (size_t)(k0 + 32 + (tid >> 6) * 8) * ldb + bn + (tid & 63), 0, 1);
            }
        } else {
            // ---- scalar guarded path (edge blocks only) ----
            #pragma unroll
            for (int i = 0; i < 16; ++i) {
                int idx = i * 256 + tid;               // 4096 A elements
                int r = idx >> 5, kk = idx & 31;
                int gm = bm + r, gk = k0 + kk;
                float v = (gm < M && gk < K) ? A[(size_t)gm * lda + gk] : 0.0f;
                As[r * PITCH + kk] = (__bf16)v;
            }
            #pragma unroll
            for (int i = 0; i < 8; ++i) {
                int idx = i * 256 + tid;               // 2048 B elements
                int n = idx >> 5, kk = idx & 31;
                int gn = bn + n, gk = k0 + kk;
                float v;
                if (TRANS_B)
                    v = (gn < N && gk < K) ? Bm[(size_t)gn * ldb + gk] : 0.0f;
                else
                    v = (gn < N && gk < K) ? Bm[(size_t)gk * ldb + gn] : 0.0f;
                Bs[n * PITCH + kk] = (__bf16)v;
            }
        }
        __syncthreads();

        FragBF af[2], bb[2];
        #pragma unroll
        for (int mi = 0; mi < 2; ++mi) {
            const int ra = (wm + mi * 16 + fr) * PITCH;
            af[mi].h[0] = *(const v8bf*)&As[ra + kA];
            af[mi].h[1] = *(const v8bf*)&As[ra + kA + 16];
        }
        #pragma unroll
        for (int ni = 0; ni < 2; ++ni) {
            const int rb = (wn + ni * 16 + fr) * PITCH;
            bb[ni].h[0] = *(const v8bf*)&Bs[rb + kB];
            bb[ni].h[1] = *(const v8bf*)&Bs[rb + kB + 16];
        }
        #pragma unroll
        for (int mi = 0; mi < 2; ++mi)
            #pragma unroll
            for (int ni = 0; ni < 2; ++ni)
                acc[mi][ni] = __builtin_amdgcn_wmma_f32_16x16x32_bf16(
                    false, af[mi].v, false, bb[ni].v,
                    (short)0, acc[mi][ni], false, false);
    }

    // --- epilogue: C/D layout — VGPR r: lanes0-15 (M=r,N=lane), lanes16-31 (M=r+8) ---
    const int nc   = lane & 15;
    const int moff = (lane < 16) ? 0 : 8;
    #pragma unroll
    for (int mi = 0; mi < 2; ++mi) {
        #pragma unroll
        for (int r = 0; r < 8; ++r) {
            int m = bm + wm + mi * 16 + moff + r;
            if (!interior && m >= M) continue;
            #pragma unroll
            for (int ni = 0; ni < 2; ++ni) {
                int n = bn + wn + ni * 16 + nc;
                if (!interior && n >= N) continue;
                float v = acc[mi][ni][r];
                if (bias) v += bias[n];
                if (DO_GELU) v = 0.5f * v * (1.0f + erff(v * 0.70710678118654752f));
                if (res) v += res[(size_t)m * ldr + n];
                C[(size_t)m * ldc + n] = v;
            }
        }
    }
}

// ---------------------------------------------------------------------------
// Host orchestration
// ---------------------------------------------------------------------------
extern "C" void kernel_launch(void* const* d_in, const int* in_sizes, int n_in,
                              void* d_out, int out_size, void* d_ws, size_t ws_size,
                              hipStream_t stream)
{
    (void)in_sizes; (void)n_in; (void)out_size; (void)ws_size;

    const int*   input_ids = (const int*)  d_in[0];
    const float* tok_emb   = (const float*)d_in[1];
    const float* pos_emb   = (const float*)d_in[2];
    const float* ln1_g     = (const float*)d_in[3];
    const float* ln1_b     = (const float*)d_in[4];
    const float* w_attn    = (const float*)d_in[5];
    const float* b_attn    = (const float*)d_in[6];
    const float* w_out     = (const float*)d_in[7];
    const float* b_out     = (const float*)d_in[8];
    const float* ln2_g     = (const float*)d_in[9];
    const float* ln2_b     = (const float*)d_in[10];
    const float* w1        = (const float*)d_in[11];
    const float* b1        = (const float*)d_in[12];
    const float* w2        = (const float*)d_in[13];
    const float* b2        = (const float*)d_in[14];
    const float* lnf_g     = (const float*)d_in[15];
    const float* lnf_b     = (const float*)d_in[16];

    // Workspace layout (floats):
    float* ws = (float*)d_ws;
    size_t off = 0;
    float* x      = ws + off; off += (size_t)MTOK * HID;        // residual stream
    float* nbuf   = ws + off; off += (size_t)MTOK * HID;        // layernorm out
    float* qkv    = ws + off; off += (size_t)MTOK * 3 * HID;    // qkv
    float* attn   = ws + off; off += (size_t)MTOK * HID;        // attention concat
    float* hbuf   = ws + off; off += (size_t)MTOK * 4 * HID;    // mlp hidden
    float* scores = ws + off; off += (size_t)NHEADS * SEQ * SEQ;// per-batch scores

    embed_kernel<<<MTOK, 256, 0, stream>>>(input_ids, tok_emb, pos_emb, x);

    const float scale = 1.0f / sqrtf((float)HD);

    for (int l = 0; l < LAYERS; ++l) {
        const float* wA  = w_attn + (size_t)l * HID * 3 * HID;
        const float* bA  = b_attn + (size_t)l * 3 * HID;
        const float* wO  = w_out  + (size_t)l * HID * HID;
        const float* bO  = b_out  + (size_t)l * HID;
        const float* w1l = w1     + (size_t)l * HID * 4 * HID;
        const float* b1l = b1     + (size_t)l * 4 * HID;
        const float* w2l = w2     + (size_t)l * 4 * HID * HID;
        const float* b2l = b2     + (size_t)l * HID;

        // n1 = LN1(x)
        layernorm_kernel<<<MTOK, 256, 0, stream>>>(
            x, HID, ln1_g + (size_t)l * HID, ln1_b + (size_t)l * HID,
            nbuf, HID, HID);

        // qkv = n1 @ wA + bA        (8192 x 2304 x 768)
        {
            dim3 g((3 * HID) / 64, MTOK / 128, 1);
            gemm_wmma_kernel<false, false><<<g, 256, 0, stream>>>(
                nbuf, 0, HID, wA, 0, 3 * HID, bA,
                nullptr, 0, 0, qkv, 0, 3 * HID, MTOK, 3 * HID, HID);
        }

        // attention, chunked per batch (scores buffer = NH*S*S = 50 MB < L2)
        for (int b = 0; b < NBATCH; ++b) {
            const float* qb = qkv + (size_t)b * SEQ * 3 * HID;
            // scores[h] = q_h @ k_h^T   (batched over heads via blockIdx.z)
            {
                dim3 g(SEQ / 64, SEQ / 128, NHEADS);
                gemm_wmma_kernel<true, false><<<g, 256, 0, stream>>>(
                    qb,            HD, 3 * HID,          // A: q, head stride HD
                    qb + HID,      HD, 3 * HID,          // B: k (NxK), used transposed
                    nullptr, nullptr, 0, 0,
                    scores, (long long)SEQ * SEQ, SEQ,
                    SEQ, SEQ, HD);
            }
            // causal softmax (mask + scale fused)
            softmax_kernel<<<NHEADS * SEQ, 256, 0, stream>>>(scores, scale);
            // attn[h] = probs @ v_h
            {
                dim3 g(HD / 64, SEQ / 128, NHEADS);
                gemm_wmma_kernel<false, false><<<g, 256, 0, stream>>>(
                    scores, (long long)SEQ * SEQ, SEQ,
                    qb + 2 * HID,  HD, 3 * HID,
                    nullptr, nullptr, 0, 0,
                    attn + (size_t)b * SEQ * HID, HD, HID,
                    SEQ, HD, SEQ);
            }
        }

        // x = x + attn @ wO + bO
        {
            dim3 g(HID / 64, MTOK / 128, 1);
            gemm_wmma_kernel<false, false><<<g, 256, 0, stream>>>(
                attn, 0, HID, wO, 0, HID, bO,
                x, 0, HID, x, 0, HID, MTOK, HID, HID);
        }

        // n2 = LN2(x)
        layernorm_kernel<<<MTOK, 256, 0, stream>>>(
            x, HID, ln2_g + (size_t)l * HID, ln2_b + (size_t)l * HID,
            nbuf, HID, HID);

        // h = gelu(n2 @ w1 + b1)
        {
            dim3 g((4 * HID) / 64, MTOK / 128, 1);
            gemm_wmma_kernel<false, true><<<g, 256, 0, stream>>>(
                nbuf, 0, HID, w1l, 0, 4 * HID, b1l,
                nullptr, 0, 0, hbuf, 0, 4 * HID, MTOK, 4 * HID, HID);
        }

        // x = x + h @ w2 + b2
        {
            dim3 g(HID / 64, MTOK / 128, 1);
            gemm_wmma_kernel<false, false><<<g, 256, 0, stream>>>(
                hbuf, 0, 4 * HID, w2l, 0, HID, b2l,
                x, 0, HID, x, 0, HID, MTOK, HID, 4 * HID);
        }
    }

    // final LN on last token of each batch -> final_encoding (d_out tail)
    float* logits = (float*)d_out;                       // (8, 50257)
    float* fe     = logits + (size_t)NBATCH * VOCAB;     // (8, 768)
    layernorm_kernel<<<NBATCH, 256, 0, stream>>>(
        x + (size_t)(SEQ - 1) * HID, (long long)SEQ * HID,
        lnf_g, lnf_b, fe, HID, HID);

    // logits = fe @ tok_emb^T   (8 x 50257 x 768)  — edge blocks use guarded path
    {
        dim3 g((VOCAB + 63) / 64, 1, 1);
        gemm_wmma_kernel<true, false><<<g, 256, 0, stream>>>(
            fe, 0, HID, tok_emb, 0, HID, nullptr,
            nullptr, 0, 0, logits, 0, VOCAB, NBATCH, VOCAB, HID);
    }
}
